// CrossAttentionAlignment_39384850104700
// MI455X (gfx1250) — matrix operand order repaired and checked by reading, not compiled, not run on hardware
//
#include <hip/hip_runtime.h>

// ---------------------------------------------------------------------------
// CrossAttentionAlignment for MI455X (gfx1250, wave32, WMMA bf16 16x16x32)
//   B=2, C=256, H=W=64 -> N=4096, heads=8, head_dim=32
//   Flash-attention style: 4096x4096 score matrices never hit HBM.
//   bf16 operands, f32 accumulation; SCALE*log2(e) folded into Wq/bq.
//   K/V tiles double-buffered with GLOBAL_LOAD_ASYNC_TO_LDS_B128 (ASYNCcnt):
//   tile kt+1 streams into LDS while tile kt is consumed by WMMAs.
// ---------------------------------------------------------------------------

#define BATCH 2
#define CDIM  256
#define NPIX  4096
#define HEADS 8
#define HD    32

typedef __attribute__((ext_vector_type(16))) __bf16 v16bf;
typedef __attribute__((ext_vector_type(8)))  float  v8f;
typedef __attribute__((ext_vector_type(4)))  __bf16 bf16x4;
typedef __attribute__((ext_vector_type(2)))  __bf16 bf16x2;

#define WMMA_BF16(a, b, c) \
  __builtin_amdgcn_wmma_f32_16x16x32_bf16(false, (a), false, (b), (short)0, (c), false, false)

// Async 16-byte global -> LDS copy (per-lane addresses), tracked by ASYNCcnt.
// LDS destination address = low 32 bits of the flat pointer.
static __device__ __forceinline__ void async_copy16(void* lds_dst, const void* gsrc) {
  asm volatile("global_load_async_to_lds_b128 %0, %1, off"
               :
               : "v"((unsigned)(unsigned long long)lds_dst), "v"(gsrc)
               : "memory");
}
static __device__ __forceinline__ void wait_async0() {   // all copies done
  asm volatile("s_wait_asynccnt 0x0" ::: "memory");
}
static __device__ __forceinline__ void wait_async2() {   // all but last 2 done
  asm volatile("s_wait_asynccnt 0x2" ::: "memory");
}

// A-matrix fragment (16x32 bf16, MxK). Per ISA 7.12.2:
//   lane(0..15)=row M, lane 16..31 = same rows, K offset +8;
//   VGPR v<4: K = 2v + {0,1}; VGPR v>=4: K = 16 + 2(v-4) + {0,1}
// Per lane this is two contiguous 16B runs -> 2x ds_load_b128.
static __device__ __forceinline__ v16bf load_a_frag(const __bf16* base, int ld) {
  const int lane = threadIdx.x & 31;
  const int row  = lane & 15;
  const int hi   = lane >> 4;
  v16bf a;
#pragma unroll
  for (int dv = 0; dv < 8; ++dv) {
    const int k = ((dv & 3) << 1) + (hi << 3) + ((dv >> 2) << 4);
    a[2 * dv]     = base[row * ld + k];
    a[2 * dv + 1] = base[row * ld + k + 1];
  }
  return a;
}

// B-matrix fragment (32x16 bf16, KxN): lane holds column N = lane%16; element s
// has K = 8*(lane/16) + (s&7) + 16*(s>>3) (same K order as the A fragment).
static __device__ __forceinline__ v16bf load_b_frag(const __bf16* base, int ld) {
  const int lane = threadIdx.x & 31;
  const int col  = lane & 15;
  const int hi   = lane >> 4;
  v16bf bv;
#pragma unroll
  for (int s = 0; s < 16; ++s) {
    const int k = (s & 7) + (hi << 3) + ((s >> 3) << 4);
    bv[s] = base[k * ld + col];
  }
  return bv;
}

static __device__ __forceinline__ bf16x4 cvt4(float4 v, float s) {
  return bf16x4{(__bf16)(v.x * s), (__bf16)(v.y * s), (__bf16)(v.z * s), (__bf16)(v.w * s)};
}

// ---------------------------------------------------------------------------
// Kernel 1: fused QKV projection.  virtual 768x256 GEMM x 4096 pixels.
//   dtile 0-3  -> Q  (scaled by 32^-0.5 * log2e), layout [b][h][n][hd]
//   dtile 4-7  -> K, layout TRANSPOSED [b][h][hd][n]  (B-frag for QK^T)
//   dtile 8-11 -> V, layout [b][h][n][hd]             (B-frag for PV)
// Region is uniform per workgroup -> uniform epilogue branches, no EXEC churn.
// ---------------------------------------------------------------------------
__global__ void __launch_bounds__(256)
qkv_proj_kernel(const float* __restrict__ dec, const float* __restrict__ enc,
                const float* __restrict__ Wq, const float* __restrict__ bq,
                const float* __restrict__ Wkv, const float* __restrict__ bkv,
                __bf16* __restrict__ Qh, __bf16* __restrict__ KTt,
                __bf16* __restrict__ Vh) {
  const int b = blockIdx.z;
  const int dtile = blockIdx.y;   // 12 tiles of 64 output channels
  const int ntile = blockIdx.x;   // 16 tiles of 256 pixels
  const int dbase = dtile * 64;
  const int nbase0 = ntile * 256;
  const bool isQ = (dtile < 4);
  const float qk = 0.17677669529663687f * 1.4426950408889634f; // 32^-0.5 * log2e
  const float wscale = isQ ? qk : 1.0f;
  const float* __restrict__ X = isQ ? dec : enc;
  const float* __restrict__ W = isQ ? Wq : Wkv;
  const int wrow0 = isQ ? dbase : (dbase - 256);

  __shared__ alignas(64) __bf16 Wt[64][32];
  __shared__ alignas(64) __bf16 Xt[32][256];

  const int tid = threadIdx.x;
  const int w = tid >> 5, lane = tid & 31;
  const int dsub = w & 3, nhalf = w >> 2;   // 4 d-subtiles x 2 n-halves

  v8f zero = {};
  v8f acc[8];
#pragma unroll
  for (int t = 0; t < 8; ++t) acc[t] = zero;

  const int rW = tid >> 2, cW = (tid & 3) * 8;          // 64x32 weight staging
  for (int ks = 0; ks < 8; ++ks) {                      // K = 256 in steps of 32
    const int kb = ks * 32;
    __syncthreads();
    {   // stage weight tile 64x32 (f32 -> bf16, scale folded for Q)
      const float* src = W + (size_t)(wrow0 + rW) * 256 + kb + cW;
      *(bf16x4*)&Wt[rW][cW]     = cvt4(*(const float4*)(src), wscale);
      *(bf16x4*)&Wt[rW][cW + 4] = cvt4(*(const float4*)(src + 4), wscale);
    }
    // stage activation tile 32x256: 8 float4 chunks per thread
#pragma unroll
    for (int i = 0; i < 8; ++i) {
      const int chunk = tid + i * 256;
      const int r = chunk >> 6;            // 0..31
      const int c = (chunk & 63) * 4;      // 0..252
      const float4 v = *(const float4*)(X + ((size_t)b * CDIM + kb + r) * NPIX + nbase0 + c);
      *(bf16x4*)&Xt[r][c] = cvt4(v, 1.0f);
    }
    __syncthreads();
    const v16bf afrag = load_a_frag(&Wt[dsub * 16][0], 32);
#pragma unroll
    for (int nt = 0; nt < 8; ++nt) {
      const v16bf bfrag = load_b_frag(&Xt[0][nhalf * 128 + nt * 16], 256);
      acc[nt] = WMMA_BF16(afrag, bfrag, acc[nt]);
    }
  }

  // epilogue (C layout: VGPR i -> row d, lane -> col n); region uniform per WG
  const int hi = lane >> 4, cl = lane & 15;
  const int drow0 = dsub * 16 + hi * 8;                  // row within 64-tile
  const float* bsrc = isQ ? (bq + dbase) : (bkv + dbase - 256);
  const float bsc = isQ ? qk : 1.0f;
  float bias8[8];
#pragma unroll
  for (int i = 0; i < 8; ++i) bias8[i] = bsrc[drow0 + i] * bsc;

  if (dtile < 4) {                                       // ---- Q, packed pairs
#pragma unroll
    for (int nt = 0; nt < 8; ++nt) {
      const int n = nbase0 + nhalf * 128 + nt * 16 + cl;
#pragma unroll
      for (int i = 0; i < 8; i += 2) {
        const int d = dbase + drow0 + i;
        const bf16x2 pv = bf16x2{(__bf16)(acc[nt][i] + bias8[i]),
                                 (__bf16)(acc[nt][i + 1] + bias8[i + 1])};
        *(bf16x2*)&Qh[(((size_t)b * HEADS + (d >> 5)) * NPIX + n) * HD + (d & 31)] = pv;
      }
    }
  } else if (dtile < 8) {                                // ---- K^T, strided
#pragma unroll
    for (int nt = 0; nt < 8; ++nt) {
      const int n = nbase0 + nhalf * 128 + nt * 16 + cl;
#pragma unroll
      for (int i = 0; i < 8; ++i) {
        const int dk = dbase - 256 + drow0 + i;
        KTt[(((size_t)b * HEADS + (dk >> 5)) * HD + (dk & 31)) * NPIX + n] =
            (__bf16)(acc[nt][i] + bias8[i]);
      }
    }
  } else {                                               // ---- V, packed pairs
#pragma unroll
    for (int nt = 0; nt < 8; ++nt) {
      const int n = nbase0 + nhalf * 128 + nt * 16 + cl;
#pragma unroll
      for (int i = 0; i < 8; i += 2) {
        const int dv = dbase - 512 + drow0 + i;
        const bf16x2 pv = bf16x2{(__bf16)(acc[nt][i] + bias8[i]),
                                 (__bf16)(acc[nt][i + 1] + bias8[i + 1])};
        *(bf16x2*)&Vh[(((size_t)b * HEADS + (dv >> 5)) * NPIX + n) * HD + (dv & 31)] = pv;
      }
    }
  }
}

// ---------------------------------------------------------------------------
// Kernel 2: flash attention.  WG = 8 waves x 16 query rows = 128 rows.
// Double-buffered async pipeline: tile kt+1 streams global->LDS (ASYNCcnt)
// while tile kt is consumed.  s_wait_asynccnt 2 = wait for current tile only
// (the 2 in-flight copies per lane belong to the next tile).
// S: 4 WMMAs (K-dim = head_dim = 32 fits one 16x16x32 instruction).
// Online softmax in exp2 domain (scale pre-folded into Q).
// P -> A-fragment via wave-private LDS strip.  PV: 4 WMMAs per tile.
// ---------------------------------------------------------------------------
__global__ void __launch_bounds__(256)
flash_attn_kernel(const __bf16* __restrict__ Qh, const __bf16* __restrict__ KTt,
                  const __bf16* __restrict__ Vh, __bf16* __restrict__ OhT) {
  const int b = blockIdx.z, h = blockIdx.y, qt = blockIdx.x;
  const int bh = b * HEADS + h;
  const int tid = threadIdx.x, w = tid >> 5, lane = tid & 31;
  const int hi = lane >> 4, cl = lane & 15;

  __shared__ alignas(64) __bf16 KTs[2][32][64];  // K^T tiles: [hd][64 keys]
  __shared__ alignas(64) __bf16 Vs[2][64][32];   // V tiles:  [64 keys][hd]
  __shared__ alignas(64) __bf16 Pb[8][16][64];   // per-wave P bounce buffer

  const int qbase = qt * 128 + w * 16;
  const v16bf qa = load_a_frag(Qh + ((size_t)bh * NPIX + qbase) * HD, HD);

  float m[8], lsum[8];
  v8f o0 = {}, o1 = {};
#pragma unroll
  for (int i = 0; i < 8; ++i) { m[i] = -1e30f; lsum[i] = 0.f; }

  const int rK = tid >> 3, cK = (tid & 7) * 8;   // 32x64 staging coords
  const int rV = tid >> 2, cV = (tid & 3) * 8;   // 64x32 staging coords
  const __bf16* kt_src = KTt + ((size_t)bh * HD + rK) * NPIX + cK;
  const __bf16* v_src  = Vh + ((size_t)bh * NPIX + rV) * HD + cV;

  // prologue: start streaming tile 0
  async_copy16(&KTs[0][rK][cK], kt_src);
  async_copy16(&Vs[0][rV][cV], v_src);

  for (int kt = 0; kt < 64; ++kt) {
    const int buf = kt & 1;
    if (kt < 63) {   // stream next tile into the other buffer (overlaps compute)
      const int kb = (kt + 1) * 64;
      async_copy16(&KTs[buf ^ 1][rK][cK], kt_src + kb);
      async_copy16(&Vs[buf ^ 1][rV][cV], v_src + (size_t)kb * HD);
      wait_async2();       // current tile's copies done; next tile in flight
    } else {
      wait_async0();
    }
    __syncthreads();       // every wave's copies for tile kt have landed

    v8f s[4];
#pragma unroll
    for (int t = 0; t < 4; ++t) {       // S = Q x K^T for 4x16 key columns
      const v16bf kf = load_b_frag(&KTs[buf][0][t * 16], 64);
      v8f z = {};
      s[t] = WMMA_BF16(qa, kf, z);
    }

    // Online softmax. C-layout: lane-group <-> row-group matches m/l/alpha.
#pragma unroll
    for (int i = 0; i < 8; ++i) {
      float mx = fmaxf(fmaxf(s[0][i], s[1][i]), fmaxf(s[2][i], s[3][i]));
#pragma unroll
      for (int off = 1; off < 16; off <<= 1) mx = fmaxf(mx, __shfl_xor(mx, off, 16));
      const float mn = fmaxf(m[i], mx);
      const float alpha = __builtin_exp2f(m[i] - mn);
      float rs = 0.f;
#pragma unroll
      for (int t = 0; t < 4; ++t) {
        const float p = __builtin_exp2f(s[t][i] - mn);
        s[t][i] = p;
        rs += p;
      }
#pragma unroll
      for (int off = 1; off < 16; off <<= 1) rs += __shfl_xor(rs, off, 16);
      lsum[i] = lsum[i] * alpha + rs;
      m[i] = mn;
      o0[i] *= alpha;
      o1[i] *= alpha;
    }

    // P (C layout) -> wave-private LDS [row][key] -> reload as A fragments.
#pragma unroll
    for (int t = 0; t < 4; ++t)
#pragma unroll
      for (int i = 0; i < 8; ++i)
        Pb[w][hi * 8 + i][t * 16 + cl] = (__bf16)s[t][i];
    __syncthreads();

#pragma unroll
    for (int half = 0; half < 2; ++half) {     // O += P x V (32 keys / frag)
      const v16bf pa = load_a_frag(&Pb[w][0][half * 32], 64);
      const v16bf v0 = load_b_frag(&Vs[buf][half * 32][0], 32);
      const v16bf v1 = load_b_frag(&Vs[buf][half * 32][16], 32);
      o0 = WMMA_BF16(pa, v0, o0);
      o1 = WMMA_BF16(pa, v1, o1);
    }
    __syncthreads();   // all waves done reading buf before it is refilled
  }

#pragma unroll
  for (int i = 0; i < 8; ++i) {
    const float inv = 1.0f / lsum[i];
    o0[i] *= inv;
    o1[i] *= inv;
  }
  // Store transposed [c_in][n] so the output-projection B-frags are contiguous.
  const size_t nbase = (size_t)qbase + hi * 8;
  const size_t row0 = ((size_t)b * CDIM + h * HD + cl) * NPIX + nbase;
  const size_t row1 = ((size_t)b * CDIM + h * HD + 16 + cl) * NPIX + nbase;
#pragma unroll
  for (int i = 0; i < 8; i += 2) {
    *(bf16x2*)(OhT + row0 + i) = bf16x2{(__bf16)o0[i], (__bf16)o0[i + 1]};
    *(bf16x2*)(OhT + row1 + i) = bf16x2{(__bf16)o1[i], (__bf16)o1[i + 1]};
  }
}

// ---------------------------------------------------------------------------
// Kernel 3: output projection  out[b][d][n] = bo[d] + sum_c Wo[d][c]*attn[c][n]
// Activation tile is already bf16 -> staged with async global->LDS copies.
// ---------------------------------------------------------------------------
__global__ void __launch_bounds__(256)
out_proj_kernel(const __bf16* __restrict__ OhT, const float* __restrict__ Wo,
                const float* __restrict__ bo, float* __restrict__ out) {
  const int b = blockIdx.z, dtile = blockIdx.y, ntile = blockIdx.x;
  const int dbase = dtile * 64, nbase0 = ntile * 256;

  __shared__ alignas(64) __bf16 Wt[64][32];
  __shared__ alignas(64) __bf16 Xt[32][256];

  const int tid = threadIdx.x, w = tid >> 5, lane = tid & 31;
  const int dsub = w & 3, nhalf = w >> 2;

  v8f zero = {};
  v8f acc[8];
#pragma unroll
  for (int t = 0; t < 8; ++t) acc[t] = zero;

  const int rW = tid >> 2, cW = (tid & 3) * 8;
  for (int ks = 0; ks < 8; ++ks) {
    const int kb = ks * 32;
    __syncthreads();
    {   // weight tile: f32 -> bf16 through VGPRs
      const float* src = Wo + (size_t)(dbase + rW) * 256 + kb + cW;
      *(bf16x4*)&Wt[rW][cW]     = cvt4(*(const float4*)(src), 1.0f);
      *(bf16x4*)&Wt[rW][cW + 4] = cvt4(*(const float4*)(src + 4), 1.0f);
    }
    // activation tile 32x256 bf16: 4 async 16B chunks per thread
#pragma unroll
    for (int i = 0; i < 4; ++i) {
      const int chunk = tid + i * 256;
      const int r = chunk >> 5;            // 0..31
      const int c = (chunk & 31) * 8;      // 0..248
      async_copy16(&Xt[r][c], OhT + ((size_t)b * CDIM + kb + r) * NPIX + nbase0 + c);
    }
    wait_async0();
    __syncthreads();
    const v16bf afrag = load_a_frag(&Wt[dsub * 16][0], 32);
#pragma unroll
    for (int nt = 0; nt < 8; ++nt) {
      const v16bf bfrag = load_b_frag(&Xt[0][nhalf * 128 + nt * 16], 256);
      acc[nt] = WMMA_BF16(afrag, bfrag, acc[nt]);
    }
  }

  const int hi = lane >> 4, cl = lane & 15;
  float bias8[8];
#pragma unroll
  for (int i = 0; i < 8; ++i) bias8[i] = bo[dbase + dsub * 16 + hi * 8 + i];
#pragma unroll
  for (int nt = 0; nt < 8; ++nt) {
    const int n = nbase0 + nhalf * 128 + nt * 16 + cl;
#pragma unroll
    for (int i = 0; i < 8; ++i) {
      const int d = dbase + dsub * 16 + hi * 8 + i;
      out[((size_t)b * CDIM + d) * NPIX + n] = acc[nt][i] + bias8[i];
    }
  }
}

// ---------------------------------------------------------------------------
extern "C" void kernel_launch(void* const* d_in, const int* in_sizes, int n_in,
                              void* d_out, int out_size, void* d_ws, size_t ws_size,
                              hipStream_t stream) {
  (void)in_sizes; (void)n_in; (void)out_size; (void)ws_size;
  const float* dec = (const float*)d_in[0];
  const float* enc = (const float*)d_in[1];
  const float* Wq  = (const float*)d_in[2];
  const float* bq  = (const float*)d_in[3];
  const float* Wkv = (const float*)d_in[4];
  const float* bkv = (const float*)d_in[5];
  const float* Wo  = (const float*)d_in[6];
  const float* bo  = (const float*)d_in[7];
  float* out = (float*)d_out;

  // Workspace (bf16): Q 4MB | K^T 4MB | V 4MB | attn-out^T 4MB = 16MB total
  char* ws = (char*)d_ws;
  __bf16* Qh  = (__bf16*)(ws);
  __bf16* KTt = (__bf16*)(ws + (4u << 20));
  __bf16* Vh  = (__bf16*)(ws + (8u << 20));
  __bf16* OhT = (__bf16*)(ws + (12u << 20));

  qkv_proj_kernel<<<dim3(16, 12, BATCH), 256, 0, stream>>>(dec, enc, Wq, bq, Wkv, bkv,
                                                           Qh, KTt, Vh);
  flash_attn_kernel<<<dim3(32, HEADS, BATCH), 256, 0, stream>>>(Qh, KTt, Vh, OhT);
  out_proj_kernel<<<dim3(16, 4, BATCH), 256, 0, stream>>>(OhT, Wo, bo, out);
}
